// HungarianMatcher_48825188221597
// MI455X (gfx1250) — matrix operand order repaired and checked by reading, not compiled, not run on hardware
//
#include <hip/hip_runtime.h>

typedef __attribute__((ext_vector_type(16))) __bf16 v16bf;
typedef __attribute__((ext_vector_type(8)))  float  v8f;

#define QN      400
#define PN      200000
#define NI      100
#define NPAD    112
#define NT      7          // 7 n-tiles of 16 = 112 columns
#define QT      25         // 25 q-tiles of 16 = 400 queries
#define SPLIT   32         // P split across grid.x
#define NCHUNK  (PN/32)    // 6250 K-chunks of 32 points (exact)
#define WAVES   8
#define SLOTS   (SPLIT*WAVES)

#define LOG2E   1.4426950408889634f
#define LN2     0.6931471805599453f

// sched_barrier mask: allow SALU(0x4) | VMEM(0x10|0x20|0x40) | TRANS(0x400)
// to cross; keep VALU (B-tile builds) and WMMA clustered per tile-group so at
// most ~2 one-hot B tiles are live at once (VGPR pressure < 256).
#define SCHED_FENCE() __builtin_amdgcn_sched_barrier(0x474)

// Stable sigmoid / log-sigmoid focal terms using raw v_exp_f32 / v_log_f32 /
// v_rcp_f32 (argument 1+exp(-|x|) is in (1,2], so no denorm guards needed):
//   p        = sigmoid(x)
//   log p    = min(x,0)  - log(1+exp(-|x|))
//   log(1-p) = min(-x,0) - log(1+exp(-|x|))
//   f1 = -0.25*(1-p)^2*log p ; f0 = -0.75*p^2*log(1-p) ; d = f1-f0
__device__ __forceinline__ void point_math(float x, float& p, float& d, float& f0)
{
    float ax  = __builtin_fabsf(x);
    float e   = __builtin_amdgcn_exp2f(ax * -LOG2E);   // exp(-|x|), raw v_exp_f32
    float opp = 1.0f + e;
    float r   = __builtin_amdgcn_rcpf(opp);            // raw v_rcp_f32
    p = (x >= 0.0f) ? r : e * r;
    float l1p = __builtin_amdgcn_logf(opp) * LN2;      // log(1+e), raw v_log_f32
    float log_p   = fminf(x,  0.0f) - l1p;
    float log_1mp = fminf(-x, 0.0f) - l1p;
    float omp = 1.0f - p;
    float f1 = -0.25f * (omp * omp) * log_p;
    f0       = -0.75f * (p * p) * log_1mp;
    d = f1 - f0;
}

__global__ void zero_counts(float* cnt)
{
    if (threadIdx.x < NPAD) cnt[threadIdx.x] = 0.0f;
}

__global__ void hist_kernel(const int* __restrict__ labels, float* __restrict__ cnt)
{
    __shared__ int h[NPAD];
    int tid = threadIdx.x;
    if (tid < NPAD) h[tid] = 0;
    __syncthreads();
    for (int i = blockIdx.x * blockDim.x + tid; i < PN; i += gridDim.x * blockDim.x)
        atomicAdd(&h[labels[i]], 1);
    __syncthreads();
    if (tid < NPAD && h[tid] != 0) atomicAdd(&cnt[tid], (float)h[tid]);
}

// Pass 1: per (split, q-tile) workgroup, accumulate
//   S1[q,j] = sum_p (f1-f0)[q,p] * T[p,j]   (bf16 hi+lo WMMA, f32 accum)
//   S2[q,j] = sum_p p[q,p]       * T[p,j]   (bf16 hi WMMA; p>0, errors average out)
//   rows[q] = (sum_p f0[q,p], sum_p p[q,p]) (full f32)
__global__ __launch_bounds__(256)
void matcher_pass1(const float* __restrict__ pred, const int* __restrict__ labels,
                   float* __restrict__ G1, float* __restrict__ G2, float* __restrict__ GR)
{
    __shared__ float sacc[2 * 16 * NPAD + 32];   // S1 | S2 | rows[16][2]
    const int tid  = threadIdx.x;
    const int lane = tid & 31;
    const int wid  = tid >> 5;
    const int hf   = lane >> 4;       // half-wave select (ISA 16-bit A/B layout)
    const int ln   = lane & 15;
    const int split = blockIdx.x;
    const int qtile = blockIdx.y;
    const int q0    = qtile * 16;

    for (int i = tid; i < 2 * 16 * NPAD + 32; i += 256) sacc[i] = 0.0f;
    __syncthreads();

    v8f C1[NT], C2[NT];
    const v8f vz = {0.f,0.f,0.f,0.f,0.f,0.f,0.f,0.f};
#pragma unroll
    for (int t = 0; t < NT; t++) { C1[t] = vz; C2[t] = vz; }

    float sum_f0 = 0.0f, sum_p = 0.0f;

    // A-matrix 16-bit layout (16x32, M=ln for both halves):
    //   element e -> K = (e<8 ? e : e+8) + hf*8  => two 32B runs per lane
    const float* rowptr = pred + (size_t)(q0 + ln) * PN;
    const int slot = split * WAVES + wid;

    for (int c = slot; c < NCHUNK; c += SLOTS) {
        const int p0 = c * 32;
        const float* ra = rowptr + p0 + hf * 8;
        __builtin_prefetch(rowptr + p0 + SLOTS * 32, 0, 1);   // global_prefetch_b8
        float4 a0 = *(const float4*)(ra);
        float4 a1 = *(const float4*)(ra + 4);
        float4 a2 = *(const float4*)(ra + 16);
        float4 a3 = *(const float4*)(ra + 20);
        float xv[16] = {a0.x,a0.y,a0.z,a0.w, a1.x,a1.y,a1.z,a1.w,
                        a2.x,a2.y,a2.z,a2.w, a3.x,a3.y,a3.z,a3.w};

        v16bf A1h, A1l, A2h;
#pragma unroll
        for (int e = 0; e < 16; e++) {
            float p, d, f0;
            point_math(xv[e], p, d, f0);
            sum_f0 += f0;
            sum_p  += p;
            __bf16 dh = (__bf16)d;
            float drm = d - (float)dh;       // -> v_fma_mix_f32_bf16
            A1h[e] = dh;
            A1l[e] = (__bf16)drm;
            A2h[e] = (__bf16)p;
        }

        // B-matrix 16-bit layout (32x16): element e -> K = (e<8 ? e : e+8) + hf*8, N = ln
        const int* lp = labels + p0 + hf * 8;
        int4 l0 = *(const int4*)(lp);
        int4 l1 = *(const int4*)(lp + 4);
        int4 l2 = *(const int4*)(lp + 16);
        int4 l3 = *(const int4*)(lp + 20);
        const int li[16] = {l0.x,l0.y,l0.z,l0.w, l1.x,l1.y,l1.z,l1.w,
                            l2.x,l2.y,l2.z,l2.w, l3.x,l3.y,l3.z,l3.w};
        unsigned short lv[16];
#pragma unroll
        for (int e = 0; e < 16; e++) lv[e] = (unsigned short)li[e];
        const unsigned short jbase = (unsigned short)ln;

        // n-tile groups {0,1}{2,3}{4,5}{6}: keep at most 2 one-hot B tiles live
#pragma unroll
        for (int tg = 0; tg < 4; tg++) {
            const int tlo = tg * 2;
            const int thi = (tg < 3) ? tlo + 2 : NT;
#pragma unroll
            for (int t = tlo; t < thi; t++) {
                const unsigned short jc = (unsigned short)(t * 16) + jbase;
                v16bf B;
#pragma unroll
                for (int e = 0; e < 16; e++)
                    B[e] = (lv[e] == jc) ? (__bf16)1.0f : (__bf16)0.0f;
                C1[t] = __builtin_amdgcn_wmma_f32_16x16x32_bf16(false, A1h, false, B, (short)0, C1[t], false, false);
                C1[t] = __builtin_amdgcn_wmma_f32_16x16x32_bf16(false, A1l, false, B, (short)0, C1[t], false, false);
                C2[t] = __builtin_amdgcn_wmma_f32_16x16x32_bf16(false, A2h, false, B, (short)0, C2[t], false, false);
            }
            SCHED_FENCE();
        }
    }

    // C/D layout: lane, vgpr r -> M = r + hf*8, N = ln
#pragma unroll
    for (int t = 0; t < NT; t++) {
#pragma unroll
        for (int r = 0; r < 8; r++) {
            int m  = r + hf * 8;
            int jj = t * 16 + ln;
            atomicAdd(&sacc[m * NPAD + jj],             C1[t][r]);
            atomicAdd(&sacc[16 * NPAD + m * NPAD + jj], C2[t][r]);
        }
    }
    atomicAdd(&sacc[2 * 16 * NPAD + ln * 2 + 0], sum_f0);
    atomicAdd(&sacc[2 * 16 * NPAD + ln * 2 + 1], sum_p);
    __syncthreads();

    const size_t wg = (size_t)split * QT + qtile;
    float* g1 = G1 + wg * (16 * NPAD);
    float* g2 = G2 + wg * (16 * NPAD);
    float* gr = GR + wg * 32;
    for (int i = tid; i < 16 * NPAD; i += 256) {
        g1[i] = sacc[i];
        g2[i] = sacc[16 * NPAD + i];
    }
    if (tid < 32) gr[tid] = sacc[2 * 16 * NPAD + tid];
}

__global__ void matcher_finalize(const float* __restrict__ G1, const float* __restrict__ G2,
                                 const float* __restrict__ GR, const float* __restrict__ CNT,
                                 float* __restrict__ out)
{
    int id = blockIdx.x * blockDim.x + threadIdx.x;
    if (id >= QN * NI) return;
    int q = id / NI, j = id % NI;
    int qt = q >> 4, m = q & 15;
    float s1 = 0.f, s2 = 0.f, rf = 0.f, rp = 0.f;
    for (int s = 0; s < SPLIT; s++) {
        size_t b  = ((size_t)s * QT + qt) * (16 * NPAD) + (size_t)m * NPAD + j;
        s1 += G1[b];
        s2 += G2[b];
        size_t rb = ((size_t)s * QT + qt) * 32 + (size_t)m * 2;
        rf += GR[rb];
        rp += GR[rb + 1];
    }
    float cost_mask = (s1 + rf) * (1.0f / (float)PN);
    float cost_dice = 1.0f - (2.0f * s2 + 1.0f) / (rp + CNT[j] + 1.0f);
    out[id] = cost_mask + cost_dice;
}

extern "C" void kernel_launch(void* const* d_in, const int* in_sizes, int n_in,
                              void* d_out, int out_size, void* d_ws, size_t ws_size,
                              hipStream_t stream)
{
    (void)in_sizes; (void)n_in; (void)out_size; (void)ws_size;
    const float* pred   = (const float*)d_in[0];
    const int*   labels = (const int*)d_in[1];
    float* out = (float*)d_out;

    float* ws  = (float*)d_ws;
    float* G1  = ws;                                        // [SPLIT][QT][16][112]
    float* G2  = G1 + (size_t)SPLIT * QT * 16 * NPAD;       // same size
    float* GR  = G2 + (size_t)SPLIT * QT * 16 * NPAD;       // [SPLIT][QT][16][2]
    float* CNT = GR + (size_t)SPLIT * QT * 32;              // [112]

    zero_counts<<<1, 128, 0, stream>>>(CNT);
    hist_kernel<<<200, 256, 0, stream>>>(labels, CNT);
    matcher_pass1<<<dim3(SPLIT, QT), dim3(256), 0, stream>>>(pred, labels, G1, G2, GR);
    int nout = QN * NI;
    matcher_finalize<<<(nout + 255) / 256, 256, 0, stream>>>(G1, G2, GR, CNT, out);
}
